// MultiHeadTimeAwareAttention_74826920231001
// MI455X (gfx1250) — matrix operand order repaired and checked by reading, not compiled, not run on hardware
//
#include <hip/hip_runtime.h>
#include <hip/hip_bf16.h>

// ---------------------------------------------------------------------------
// CDNA5 (gfx1250) helpers
// ---------------------------------------------------------------------------
typedef __attribute__((ext_vector_type(16))) __bf16 v16bf;
typedef __attribute__((ext_vector_type(2)))  __bf16 v2bf;
typedef __attribute__((ext_vector_type(8)))  float  v8f;

__device__ __forceinline__ unsigned int f2bf(float f) {
  unsigned int u = __float_as_uint(f);
  u += 0x7FFFu + ((u >> 16) & 1u);  // round-to-nearest-even
  return u >> 16;
}
// pack two fp32 -> two bf16 in one dword (single v_cvt_pk_bf16_f32 if available)
__device__ __forceinline__ unsigned int pack2(float lo, float hi) {
#if __has_builtin(__builtin_amdgcn_cvt_pk_bf16_f32)
  union { v2bf v; unsigned int u; } x;
  x.v = __builtin_amdgcn_cvt_pk_bf16_f32(lo, hi);
  return x.u;
#else
  return f2bf(lo) | (f2bf(hi) << 16);
#endif
}
__device__ __forceinline__ unsigned short bf1(float v) {
  return (unsigned short)pack2(v, v);
}

union Frag16 { uint4 q[2]; unsigned int u[8]; v16bf v; };

// A-matrix fragment 16x32 bf16 from row-major LDS tile.
// lane half=0 -> k in [0,8)+[16,24); half=1 -> [8,16)+[24,32)
__device__ __forceinline__ v16bf load_fragA(const unsigned short* base, int stride,
                                            int row, int k0, int half) {
  const unsigned short* p = base + row * stride + k0 + half * 8;
  Frag16 x;
  x.q[0] = *(const uint4*)(p);
  x.q[1] = *(const uint4*)(p + 16);
  return x.v;
}
// B-matrix fragment 32x16 bf16; source stored [n][k] row-major in LDS.
__device__ __forceinline__ v16bf load_fragB(const unsigned short* base, int stride,
                                            int ncol, int k0, int half) {
  const unsigned short* p = base + ncol * stride + k0 + half * 8;
  Frag16 x;
  x.q[0] = *(const uint4*)(p);
  x.q[1] = *(const uint4*)(p + 16);
  return x.v;
}

#define WMMA_BF16(A, B, C) \
  __builtin_amdgcn_wmma_f32_16x16x32_bf16(false, (A), false, (B), (short)0, (C), false, false)

// CDNA5 async global->LDS copy, 16B per lane, tracked by ASYNCcnt.
__device__ __forceinline__ void async_load_b128(unsigned lds_off, unsigned gl_off,
                                                const void* base) {
  asm volatile("global_load_async_to_lds_b128 %0, %1, %2"
               :: "v"(lds_off), "v"(gl_off), "s"(base) : "memory");
}
__device__ __forceinline__ void wait_async() {
  asm volatile("s_wait_asynccnt 0x0" ::: "memory");
}
__device__ __forceinline__ unsigned lds_base_off(const void* shared_ptr) {
  return (unsigned)(unsigned long long)shared_ptr;  // addr[31:0] = LDS offset
}

// 16-lane XOR reductions via ds_swizzle (group-of-32 mode, and_mask=0x1f)
template <int MASK>
__device__ __forceinline__ float swz_xor(float v) {
  return __int_as_float(
      __builtin_amdgcn_ds_swizzle(__float_as_int(v), 0x1F | (MASK << 10)));
}
__device__ __forceinline__ float redmax16(float v) {
  v = fmaxf(v, swz_xor<1>(v));
  v = fmaxf(v, swz_xor<2>(v));
  v = fmaxf(v, swz_xor<4>(v));
  v = fmaxf(v, swz_xor<8>(v));
  return v;
}
__device__ __forceinline__ float redsum16(float v) {
  v += swz_xor<1>(v);
  v += swz_xor<2>(v);
  v += swz_xor<4>(v);
  v += swz_xor<8>(v);
  return v;
}

// ---------------------------------------------------------------------------
// fp32 -> bf16 bulk converter (n must be a multiple of 4)
// ---------------------------------------------------------------------------
__global__ void cvt_f32_bf16(const float* __restrict__ in,
                             unsigned short* __restrict__ out, long long n4) {
  long long i = blockIdx.x * (long long)blockDim.x + threadIdx.x;
  long long stride = (long long)gridDim.x * blockDim.x;
  for (; i < n4; i += stride) {
    float4 f = ((const float4*)in)[i];
    ((uint2*)out)[i] = make_uint2(pack2(f.x, f.y), pack2(f.z, f.w));
  }
}

// ---------------------------------------------------------------------------
// GEMM: C[M,N] = A[M,K]bf16 @ W[N,K]bf16^T + bias[N] (+ resid[M,N] fp32)
// Block tile 128x64x64, 256 threads = 8 waves, wave tile 32x32 (4 WMMA accs).
// Double-buffered GLOBAL_LOAD_ASYNC_TO_LDS_B128 pipeline: next tile's DMA
// runs underneath the current tile's WMMAs.
// ---------------------------------------------------------------------------
#define BM 128
#define BN 64
#define BK 64
#define TST (BK + 8)  // u16 stride; 144B rows (16B aligned)

template <bool BF16_OUT>
__global__ __launch_bounds__(256) void gemm_bf16_wmma(
    const unsigned short* __restrict__ A, const unsigned short* __restrict__ W,
    const float* __restrict__ bias, const float* __restrict__ resid,
    void* __restrict__ Cout, int M, int N, int K) {
  __shared__ unsigned short sA[2][BM * TST];
  __shared__ unsigned short sB[2][BN * TST];
  const int tid  = threadIdx.x;
  const int wave = tid >> 5, lane = tid & 31;
  const int half = lane >> 4, l16 = lane & 15;
  const int mi = wave & 3, ni = wave >> 2;  // 4x2 waves: 32-row x 32-col tiles
  const int m0 = blockIdx.y * BM, n0 = blockIdx.x * BN;

  // Precompute per-thread staging addresses (A: 4x16B, B: 2x16B per thread)
  unsigned ldsA[2][4], gofA[4], ldsB[2][2], gofB[2];
#pragma unroll
  for (int i = 0; i < 4; ++i) {
    int idx = tid + i * 256;
    int r = idx >> 3, c8 = idx & 7;  // 8 x 16B per 64-elem row
    gofA[i] = (unsigned)(((m0 + r) * K) * 2 + c8 * 16);
    ldsA[0][i] = lds_base_off(sA[0]) + r * (TST * 2) + c8 * 16;
    ldsA[1][i] = lds_base_off(sA[1]) + r * (TST * 2) + c8 * 16;
  }
#pragma unroll
  for (int i = 0; i < 2; ++i) {
    int idx = tid + i * 256;
    int r = idx >> 3, c8 = idx & 7;
    gofB[i] = (unsigned)(((n0 + r) * K) * 2 + c8 * 16);
    ldsB[0][i] = lds_base_off(sB[0]) + r * (TST * 2) + c8 * 16;
    ldsB[1][i] = lds_base_off(sB[1]) + r * (TST * 2) + c8 * 16;
  }
  auto issue = [&](int k0, int b) {
#pragma unroll
    for (int i = 0; i < 4; ++i) async_load_b128(ldsA[b][i], gofA[i] + k0 * 2, A);
#pragma unroll
    for (int i = 0; i < 2; ++i) async_load_b128(ldsB[b][i], gofB[i] + k0 * 2, W);
  };

  v8f a00 = {}, a01 = {}, a10 = {}, a11 = {};
  issue(0, 0);
  wait_async();
  __syncthreads();
  int cur = 0;
  for (int k0 = 0; k0 < K; k0 += BK) {
    if (k0 + BK < K) issue(k0 + BK, cur ^ 1);  // DMA next tile under WMMAs
    const unsigned short* pA = sA[cur];
    const unsigned short* pB = sB[cur];
#pragma unroll
    for (int ks = 0; ks < BK; ks += 32) {
      v16bf fa0 = load_fragA(pA, TST, mi * 32 + l16, ks, half);
      v16bf fa1 = load_fragA(pA, TST, mi * 32 + 16 + l16, ks, half);
      v16bf fb0 = load_fragB(pB, TST, ni * 32 + l16, ks, half);
      v16bf fb1 = load_fragB(pB, TST, ni * 32 + 16 + l16, ks, half);
      a00 = WMMA_BF16(fa0, fb0, a00);
      a01 = WMMA_BF16(fa0, fb1, a01);
      a10 = WMMA_BF16(fa1, fb0, a10);
      a11 = WMMA_BF16(fa1, fb1, a11);
    }
    wait_async();
    __syncthreads();
    cur ^= 1;
  }

  // D layout: VGPR r -> row (r + 8*half), col = l16
  const int nA = n0 + ni * 32 + l16;
  const int nB = nA + 16;
  const float biasA = bias[nA], biasB = bias[nB];
#pragma unroll
  for (int fm = 0; fm < 2; ++fm) {
    v8f* accA = fm ? &a10 : &a00;
    v8f* accB = fm ? &a11 : &a01;
#pragma unroll
    for (int r = 0; r < 8; ++r) {
      int m = m0 + mi * 32 + fm * 16 + half * 8 + r;
      size_t ro = (size_t)m * N;
      float va = (*accA)[r] + biasA;
      float vb = (*accB)[r] + biasB;
      if (BF16_OUT) {
        unsigned short* Cb = (unsigned short*)Cout;
        Cb[ro + nA] = bf1(va);
        Cb[ro + nB] = bf1(vb);
      } else {
        float* Cf = (float*)Cout;
        if (resid) { va += resid[ro + nA]; vb += resid[ro + nB]; }
        Cf[ro + nA] = va;
        Cf[ro + nB] = vb;
      }
    }
  }
}

// ---------------------------------------------------------------------------
// time_bias[h][s] = time_emb[s,:] . tp_w[h,:] + tp_b[h]   (H*S dots, tiny)
// ---------------------------------------------------------------------------
__global__ void time_bias_kernel(const float* __restrict__ temb,
                                 const float* __restrict__ tpw,
                                 const float* __restrict__ tpb,
                                 float* __restrict__ out, int S, int E, int H) {
  int idx = blockIdx.x * blockDim.x + threadIdx.x;
  if (idx >= H * S) return;
  int h = idx / S, s = idx - h * S;
  const float4* a = (const float4*)(temb + (size_t)s * E);
  const float4* w = (const float4*)(tpw + (size_t)h * E);
  float acc = 0.f;
  for (int i = 0; i < E / 4; ++i) {
    float4 x = a[i], y = w[i];
    acc += x.x * y.x + x.y * y.y + x.z * y.z + x.w * y.w;
  }
  out[idx] = acc + tpb[h];
}

// ---------------------------------------------------------------------------
// Flash attention: bf16 Q/K/V, one head per blockIdx.y, 64 query rows/block
// (4 waves x 16 rows), key tiles of 32, Dh = 128. Double-buffered tiles:
// K via async-to-LDS, V global loads issued early (land during score WMMAs).
// ---------------------------------------------------------------------------
#define TS  32
#define KST (128 + 8)  // sK u16 stride (272B rows, 16B aligned)
#define VST (TS + 8)   // sV u16 stride
#define PST (TS + 8)

__global__ __launch_bounds__(128) void attn_flash_wmma(
    const unsigned short* __restrict__ Qb, const unsigned short* __restrict__ Kb,
    const unsigned short* __restrict__ Vb, const float* __restrict__ tbias,
    float* __restrict__ Cx, int L, int S, int H) {
  __shared__ unsigned short sK[2][TS * KST];   // [s][d] bf16
  __shared__ unsigned short sV[2][128 * VST];  // [d][s] bf16 (transposed)
  __shared__ unsigned short sP[4][16 * PST];   // per-wave probs [m][s]
  const int E = H * 128;
  const int h = blockIdx.y;
  const int q0 = blockIdx.x * 64;
  const int tid = threadIdx.x;
  const int wave = tid >> 5, lane = tid & 31;
  const int half = lane >> 4, l16 = lane & 15;
  const float scale = 0.088388347648318447f;  // 1/sqrt(128)

  // per-thread staging addresses (K: 4x16B async, V: 4x16B loads per thread)
  const int str = (tid + 0) >> 4;          // tile row handled by this thread
  const int stc8 = tid & 15;               // 16B column within 128-elem row
  unsigned ldsK[2][4], gofK[4];
#pragma unroll
  for (int i = 0; i < 4; ++i) {
    int r = str + i * 8;                   // rows str, str+8, +16, +24
    gofK[i] = (unsigned)((r * E + h * 128) * 2 + stc8 * 16);
    ldsK[0][i] = lds_base_off(sK[0]) + r * (KST * 2) + stc8 * 16;
    ldsK[1][i] = lds_base_off(sK[1]) + r * (KST * 2) + stc8 * 16;
  }

  // Q fragments: 16 rows x 128 cols -> 4 A-fragments straight from global bf16
  v16bf qf[4];
  {
    const unsigned short* qp = Qb + (size_t)(q0 + wave * 16 + l16) * E + h * 128;
#pragma unroll
    for (int kk = 0; kk < 4; ++kk) {
      Frag16 f;
      f.q[0] = *(const uint4*)(qp + kk * 32 + half * 8);
      f.q[1] = *(const uint4*)(qp + kk * 32 + 16 + half * 8);
      qf[kk] = f.v;
    }
  }

  float rmax[8], rsum[8];
  v8f cacc[8];
  v8f vzero = {};
#pragma unroll
  for (int r = 0; r < 8; ++r) { rmax[r] = -3.0e38f; rsum[r] = 0.0f; }
#pragma unroll
  for (int i = 0; i < 8; ++i) cacc[i] = vzero;

  const float* tbh = tbias + (size_t)h * S;

  auto issueK = [&](int s0, int b) {
    unsigned so = (unsigned)(s0 * E * 2);
#pragma unroll
    for (int i = 0; i < 4; ++i) async_load_b128(ldsK[b][i], gofK[i] + so, Kb);
  };
  auto loadV = [&](int s0, uint4* vr) {
#pragma unroll
    for (int i = 0; i < 4; ++i)
      vr[i] = *(const uint4*)(Vb + (size_t)(s0 + str + i * 8) * E + h * 128 + stc8 * 8);
  };
  auto storeV = [&](int b, const uint4* vr) {
#pragma unroll
    for (int i = 0; i < 4; ++i) {
      const unsigned short* e = (const unsigned short*)&vr[i];
#pragma unroll
      for (int d8 = 0; d8 < 8; ++d8)
        sV[b][(stc8 * 8 + d8) * VST + (str + i * 8)] = e[d8];
    }
  };

  // prologue: stage tile 0
  {
    uint4 vr[4];
    issueK(0, 0);
    loadV(0, vr);
    storeV(0, vr);
    wait_async();
    __syncthreads();
  }

  int cur = 0;
  for (int s0 = 0; s0 < S; s0 += TS) {
    const bool has_next = (s0 + TS) < S;
    uint4 vr[4];
    if (has_next) {
      issueK(s0 + TS, cur ^ 1);   // async DMA under the WMMAs below
      loadV(s0 + TS, vr);         // loads sink to first use (after scores)
    }

    // scores = Q K^T * scale + time_bias : two 16-col D fragments
    float sc[2][8];
#pragma unroll
    for (int j = 0; j < 2; ++j) {
      v8f sacc = {};
#pragma unroll
      for (int kk = 0; kk < 4; ++kk) {
        v16bf b = load_fragB(sK[cur], KST, j * 16 + l16, kk * 32, half);
        sacc = WMMA_BF16(qf[kk], b, sacc);
      }
      float tb = tbh[s0 + j * 16 + l16];  // per-column bias (col = l16)
#pragma unroll
      for (int r = 0; r < 8; ++r) sc[j][r] = sacc[r] * scale + tb;
    }

    // online softmax; lane owns rows m = r + 8*half; reduce over 16 cols
#pragma unroll
    for (int r = 0; r < 8; ++r) {
      float tm = redmax16(fmaxf(sc[0][r], sc[1][r]));
      float mnew = fmaxf(rmax[r], tm);
      float corr = __expf(rmax[r] - mnew);
      float p0 = __expf(sc[0][r] - mnew);
      float p1 = __expf(sc[1][r] - mnew);
      rsum[r] = rsum[r] * corr + redsum16(p0 + p1);
      rmax[r] = mnew;
#pragma unroll
      for (int i = 0; i < 8; ++i) cacc[i][r] *= corr;
      unsigned pk = pack2(p0, p1);
      unsigned short* pp = sP[wave] + (half * 8 + r) * PST + l16;
      pp[0]  = (unsigned short)(pk & 0xFFFFu);
      pp[16] = (unsigned short)(pk >> 16);
    }

    // transpose-store next V tile (into buf cur^1; reads below use buf cur)
    if (has_next) storeV(cur ^ 1, vr);

    // per-wave LDS RAW fence (sP round trip; also covers sV stores ordering)
    asm volatile("s_wait_dscnt 0" ::: "memory");

    // ctx += P @ V : A = P (16x32), B = V (32x128 -> 8 fragments)
    v16bf ap = load_fragA(sP[wave], PST, l16, 0, half);
#pragma unroll
    for (int j2 = 0; j2 < 8; ++j2) {
      v16bf b = load_fragB(sV[cur], VST, j2 * 16 + l16, 0, half);
      cacc[j2] = WMMA_BF16(ap, b, cacc[j2]);
    }

    wait_async();
    __syncthreads();
    cur ^= 1;
  }

  // finalize: divide by softmax denominator, write fp32 ctx
#pragma unroll
  for (int r = 0; r < 8; ++r) {
    float inv = 1.0f / rsum[r];
    size_t ro = (size_t)(q0 + wave * 16 + half * 8 + r) * E + h * 128 + l16;
#pragma unroll
    for (int j2 = 0; j2 < 8; ++j2) Cx[ro + j2 * 16] = cacc[j2][r] * inv;
  }
}

// ---------------------------------------------------------------------------
// In-place LayerNorm over rows of X (residual already fused into X)
// ---------------------------------------------------------------------------
__global__ __launch_bounds__(256) void layernorm_kernel(
    float* __restrict__ X, const float* __restrict__ g,
    const float* __restrict__ b, int E) {
  __shared__ float rs[256], rq[256];
  const int tid = threadIdx.x;
  float* xr = X + (size_t)blockIdx.x * E;
  float v[8], ls = 0.f, lq = 0.f;
#pragma unroll
  for (int i = 0; i < 8; ++i) {
    v[i] = xr[tid + i * 256];
    ls += v[i];
    lq += v[i] * v[i];
  }
  rs[tid] = ls;
  rq[tid] = lq;
  __syncthreads();
  for (int off = 128; off > 0; off >>= 1) {
    if (tid < off) { rs[tid] += rs[tid + off]; rq[tid] += rq[tid + off]; }
    __syncthreads();
  }
  float mu  = rs[0] / (float)E;
  float var = rq[0] / (float)E - mu * mu;
  float inv = rsqrtf(var + 1e-5f);
#pragma unroll
  for (int i = 0; i < 8; ++i) {
    int e = tid + i * 256;
    xr[e] = (v[i] - mu) * inv * g[e] + b[e];
  }
}

// ---------------------------------------------------------------------------
// Launcher
// ---------------------------------------------------------------------------
extern "C" void kernel_launch(void* const* d_in, const int* in_sizes, int n_in,
                              void* d_out, int out_size, void* d_ws, size_t ws_size,
                              hipStream_t stream) {
  (void)in_sizes; (void)n_in; (void)out_size; (void)ws_size;
  const float* query = (const float*)d_in[0];   // (L,1,E)
  const float* key   = (const float*)d_in[1];   // (S,1,E)
  const float* value = (const float*)d_in[2];   // (S,1,E)
  const float* temb  = (const float*)d_in[3];   // (S,E)
  const float* in_w  = (const float*)d_in[4];   // (3E,E)
  const float* in_b  = (const float*)d_in[5];   // (3E)
  const float* out_w = (const float*)d_in[6];   // (E,E)
  const float* out_b = (const float*)d_in[7];   // (E)
  const float* tp_w  = (const float*)d_in[8];   // (H,E)
  const float* tp_b  = (const float*)d_in[9];   // (H)
  const float* ln_g  = (const float*)d_in[10];  // (E)
  const float* ln_b  = (const float*)d_in[11];  // (E)

  const int L = 2048, S = 2048, E = 2048, H = 16;

  // Workspace layout (bf16 elements unless noted)
  unsigned short* wsb     = (unsigned short*)d_ws;
  unsigned short* inw_bf  = wsb;                            // 3*E*E
  unsigned short* outw_bf = inw_bf + (size_t)3 * E * E;     // E*E
  unsigned short* qin_bf  = outw_bf + (size_t)E * E;        // L*E
  unsigned short* kin_bf  = qin_bf + (size_t)L * E;         // S*E
  unsigned short* vin_bf  = kin_bf + (size_t)S * E;         // S*E
  unsigned short* Qb      = vin_bf + (size_t)S * E;         // L*E
  unsigned short* Kb      = Qb + (size_t)L * E;             // S*E
  unsigned short* Vb      = Kb + (size_t)S * E;             // S*E
  float*          Tb      = (float*)(Vb + (size_t)S * E);   // H*S fp32
  float*          ctx     = (float*)qin_bf;  // alias: q/k/v inputs dead post-QKV
  unsigned short* ctx_bf  = Qb;              // alias: Qb dead post-attention
  float*          out     = (float*)d_out;

  // 1) bulk fp32->bf16 conversions (one pass each; weights converted once)
  cvt_f32_bf16<<<1024, 256, 0, stream>>>(query, qin_bf, (long long)L * E / 4);
  cvt_f32_bf16<<<1024, 256, 0, stream>>>(key,   kin_bf, (long long)S * E / 4);
  cvt_f32_bf16<<<1024, 256, 0, stream>>>(value, vin_bf, (long long)S * E / 4);
  cvt_f32_bf16<<<2048, 256, 0, stream>>>(in_w,  inw_bf, (long long)3 * E * E / 4);
  cvt_f32_bf16<<<1024, 256, 0, stream>>>(out_w, outw_bf, (long long)E * E / 4);

  // 2) Q/K/V projections -> bf16 outputs (bias fused)
  dim3 blk(256), grd(E / BN, L / BM);
  gemm_bf16_wmma<true><<<grd, blk, 0, stream>>>(qin_bf, inw_bf,                 in_b,         nullptr, Qb, L, E, E);
  gemm_bf16_wmma<true><<<grd, blk, 0, stream>>>(kin_bf, inw_bf + (size_t)E * E, in_b + E,     nullptr, Kb, S, E, E);
  gemm_bf16_wmma<true><<<grd, blk, 0, stream>>>(vin_bf, inw_bf + (size_t)2*E*E, in_b + 2 * E, nullptr, Vb, S, E, E);

  // 3) per-head additive time bias
  time_bias_kernel<<<(H * S) / 256, 256, 0, stream>>>(temb, tp_w, tp_b, Tb, S, E, H);

  // 4) flash attention (writes fp32 ctx), then convert ctx to bf16
  attn_flash_wmma<<<dim3(L / 64, H), 128, 0, stream>>>(Qb, Kb, Vb, Tb, ctx, L, S, H);
  cvt_f32_bf16<<<1024, 256, 0, stream>>>(ctx, ctx_bf, (long long)L * E / 4);

  // 5) output projection + residual (query) fused -> pre-LN x in d_out
  gemm_bf16_wmma<false><<<grd, blk, 0, stream>>>(ctx_bf, outw_bf, out_b, query, out, L, E, E);

  // 6) in-place LayerNorm
  layernorm_kernel<<<L, 256, 0, stream>>>(out, ln_g, ln_b, E);
}
